// Dynamic_Model_89120571392622
// MI455X (gfx1250) — matrix operand order repaired and checked by reading, not compiled, not run on hardware
//
#include <hip/hip_runtime.h>
#include <hip/hip_bf16.h>
#include <math.h>
#include <stdint.h>

// MI455X / gfx1250, wave32. All matrix work via v_wmma_f32_16x16x32_bf16.
typedef __bf16 bf16;
typedef __attribute__((ext_vector_type(16))) __bf16 v16bf;
typedef __attribute__((ext_vector_type(8)))  float  v8f;

#define WMMA_BF16(a, b, c) \
  __builtin_amdgcn_wmma_f32_16x16x32_bf16(false, (a), false, (b), (short)0, (c), false, false)

__device__ __forceinline__ float gelu_tanh(float x) {
  float x3 = x * x * x;
  return 0.5f * x * (1.0f + tanhf(0.7978845608028654f * (x + 0.044715f * x3)));
}

__device__ __forceinline__ float wave_sum(float v) {
#pragma unroll
  for (int m = 16; m >= 1; m >>= 1) v += __shfl_xor(v, m, 32);
  return v;
}

// ---------------------------------------------------------------------------
// Pack f32 weight [K,Nc] into WMMA B-fragment dword layout:
// dword index i = ((ktile*ntiles + ntile)*32 + lane)*8 + j
//   holds bf16 pair for k = ktile*32 + (j/4)*16 + (lane/16)*8 + 2*(j%4), n = ntile*16 + lane%16
// so each wave fetches one 16x32 B fragment as 2 coalesced b128 loads.
// ---------------------------------------------------------------------------
__global__ void pack_w(const float* __restrict__ W, uint32_t* __restrict__ out,
                       int K, int Nc) {
  int i = blockIdx.x * blockDim.x + threadIdx.x;
  int total = (K * Nc) >> 1;
  if (i >= total) return;
  int j = i & 7, lane = (i >> 3) & 31, tile = i >> 8;
  int ntn = Nc >> 4;
  int tk = tile / ntn, tn = tile - tk * ntn;
  int k = tk * 32 + ((j >> 2) << 4) + ((lane >> 4) << 3) + ((j & 3) << 1);
  int n = tn * 16 + (lane & 15);
  union { bf16 h[2]; uint32_t u; } c;
  c.h[0] = (bf16)W[(size_t)k * Nc + n];
  c.h[1] = (bf16)W[(size_t)(k + 1) * Nc + n];
  out[i] = c.u;
}

// ---------------------------------------------------------------------------
// LayerNorm over 128 features, one wave per token (lane handles 4 features).
// If x2 != nullptr: token stream is concat([x, x2], axis=1) per batch of
// `halfTok` tokens each (predictor input).
// ---------------------------------------------------------------------------
__global__ void ln_kernel(const float* __restrict__ x, const float* __restrict__ x2,
                          const float* __restrict__ g, const float* __restrict__ bta,
                          float* __restrict__ out, int T, int halfTok) {
  int wave = threadIdx.x >> 5, lane = threadIdx.x & 31;
  int t = blockIdx.x * (blockDim.x >> 5) + wave;
  if (t >= T) return;
  const float* src;
  if (x2) {
    int bb = t / (2 * halfTok);
    int r = t - bb * 2 * halfTok;
    src = (r < halfTok) ? (x + (size_t)(bb * halfTok + r) * 128)
                        : (x2 + (size_t)(bb * halfTok + (r - halfTok)) * 128);
  } else {
    src = x + (size_t)t * 128;
  }
  float4 v = ((const float4*)src)[lane];
  float mean = wave_sum(v.x + v.y + v.z + v.w) * (1.0f / 128.0f);
  float dx = v.x - mean, dy = v.y - mean, dz = v.z - mean, dw = v.w - mean;
  float var = wave_sum(dx * dx + dy * dy + dz * dz + dw * dw) * (1.0f / 128.0f);
  float rs = rsqrtf(var + 1e-6f);
  float4 gg = ((const float4*)g)[lane];
  float4 bb4 = ((const float4*)bta)[lane];
  float4 o;
  o.x = dx * rs * gg.x + bb4.x;
  o.y = dy * rs * gg.y + bb4.y;
  o.z = dz * rs * gg.z + bb4.z;
  o.w = dw * rs * gg.w + bb4.w;
  ((float4*)(out + (size_t)t * 128))[lane] = o;
}

// ---------------------------------------------------------------------------
// Generic WMMA GEMM: out[M,Nc] = act(A[M,K] @ W + bias) (+ resid)
// W is PACKED (see pack_w). One wave computes one 16x16 output tile.
// A 16x32 f32 tile is staged into per-wave LDS via async copies
// (global_load_async_to_lds_b128 + s_wait_asynccnt), then read as fragments.
// If glob != nullptr (predictor concat mode, K==128): direct-load fallback;
// A cols 0..63 from A (row stride K), cols 64..127 from glob[m/tokPerBatch].
// ---------------------------------------------------------------------------
__global__ void __launch_bounds__(256)
gemm_wmma(const float* __restrict__ A, const uint32_t* __restrict__ W,
          const float* __restrict__ bias, const float* __restrict__ resid,
          const float* __restrict__ glob, float* __restrict__ out,
          int M, int K, int Nc, int actGelu, int tokPerBatch) {
  const int AS = 36;  // padded row stride (floats) for the staged A tile
  __shared__ float aS[8 * 16 * AS];
  int wave = threadIdx.x >> 5, lane = threadIdx.x & 31;
  int lh = lane >> 4, l16 = lane & 15;
  int ntiles = Nc >> 4;
  int tile = blockIdx.x * (blockDim.x >> 5) + wave;
  int tiles = (M >> 4) * ntiles;
  if (tile >= tiles) return;  // uniform per-wave: EXEC stays all-ones for WMMA
  int tm = tile / ntiles, tn = tile - tm * ntiles;
  float* As = aS + wave * 16 * AS;
  v8f acc = {0.f, 0.f, 0.f, 0.f, 0.f, 0.f, 0.f, 0.f};
  for (int kk = 0; kk < K; kk += 32) {
    v16bf a, b;
    // ---- B fragment: 2 coalesced b128 loads from packed weights ----
    {
      size_t fb = (((size_t)(kk >> 5) * ntiles + tn) * 32 + lane) * 8;
      union { uint4 q[2]; v16bf v; } bu;
      bu.q[0] = *(const uint4*)(W + fb);
      bu.q[1] = *(const uint4*)(W + fb + 4);
      b = bu.v;
      if (kk + 32 < K)
        __builtin_prefetch(W + fb + (size_t)ntiles * 256, 0, 0);
    }
    // ---- A fragment ----
    if (glob) {
      // predictor concat mode: direct loads
#pragma unroll
      for (int j = 0; j < 8; ++j) {
        int ka = kk + ((j >> 2) << 4) + (lh << 3) + ((j & 3) << 1);
        int m = tm * 16 + l16;
        int bidx = m / tokPerBatch;
        float a0 = (ka < 64) ? A[(size_t)m * K + ka] : glob[bidx * 64 + (ka - 64)];
        float a1 = (ka + 1 < 64) ? A[(size_t)m * K + ka + 1] : glob[bidx * 64 + (ka + 1 - 64)];
        a[2 * j] = (bf16)a0;
        a[2 * j + 1] = (bf16)a1;
      }
    } else {
      // stage 16x32 f32 A tile into LDS with async copies (16B per lane-issue)
#pragma unroll
      for (int it = 0; it < 4; ++it) {
        int cid = it * 32 + lane;  // 0..127 : 16 rows x 8 chunks of 16B
        int row = cid >> 3;
        int c = cid & 7;
        const float* ga = A + (size_t)(tm * 16 + row) * K + kk + c * 4;
        uint32_t ldsOff = (uint32_t)(size_t)(As + row * AS + c * 4);
        unsigned long long gaddr = (unsigned long long)(size_t)ga;
        asm volatile("global_load_async_to_lds_b128 %0, %1, off"
                     :: "v"(ldsOff), "v"(gaddr) : "memory");
      }
      asm volatile("s_wait_asynccnt 0" ::: "memory");
#pragma unroll
      for (int j = 0; j < 8; ++j) {
        int ka = ((j >> 2) << 4) + (lh << 3) + ((j & 3) << 1);  // ISA A 16x32 layout
        const float* ap = As + l16 * AS + ka;
        a[2 * j] = (bf16)ap[0];
        a[2 * j + 1] = (bf16)ap[1];
      }
      asm volatile("s_wait_dscnt 0" ::: "memory");  // reads done before next stage
    }
    acc = WMMA_BF16(a, b, acc);
  }
#pragma unroll
  for (int r = 0; r < 8; ++r) {
    int n = tn * 16 + l16;
    int mr = tm * 16 + r + (lh << 3);  // ISA C/D layout: m = r + half*8
    float v = acc[r] + (bias ? bias[n] : 0.0f);
    if (actGelu) v = gelu_tanh(v);
    if (resid) v += resid[(size_t)mr * Nc + n];
    out[(size_t)mr * Nc + n] = v;
  }
}

// ---------------------------------------------------------------------------
// Per-batch mean of predictor hidden cols 64..127 (policy == ones).
// grid = B*64 blocks of 256 threads.
// ---------------------------------------------------------------------------
__global__ void glob_reduce(const float* __restrict__ h, float* __restrict__ glob) {
  int b = blockIdx.x >> 6;
  int f = blockIdx.x & 63;
  __shared__ float red[256];
  float s = 0.f;
  for (int t = threadIdx.x; t < 4608; t += 256)
    s += h[((size_t)(b * 4608 + t)) * 128 + 64 + f];
  red[threadIdx.x] = s;
  __syncthreads();
  for (int st = 128; st > 0; st >>= 1) {
    if (threadIdx.x < st) red[threadIdx.x] += red[threadIdx.x + st];
    __syncthreads();
  }
  if (threadIdx.x == 0) glob[b * 64 + f] = red[0] * (1.0f / 4608.0f);
}

// ---------------------------------------------------------------------------
// Predictor tail: h1[t,64] -> gelu(@o2 64->32) -> @o3 32->2 -> log_softmax
// -> + Gumbel noise -> argmax -> policy (1.0 if class 0 wins).
// ---------------------------------------------------------------------------
__global__ void pred_tail(const float* __restrict__ h1, const float* __restrict__ o2w,
                          const float* __restrict__ o2b, const float* __restrict__ o3w,
                          const float* __restrict__ o3b, const float* __restrict__ u,
                          float* __restrict__ pol) {
  int t = blockIdx.x * blockDim.x + threadIdx.x;
  if (t >= 18432) return;
  const float* hr = h1 + (size_t)t * 64;
  float h2[32];
#pragma unroll 4
  for (int j = 0; j < 32; ++j) {
    float a = o2b[j];
    for (int i = 0; i < 64; ++i) a += hr[i] * o2w[i * 32 + j];
    h2[j] = gelu_tanh(a);
  }
  float z0 = o3b[0], z1 = o3b[1];
  for (int j = 0; j < 32; ++j) {
    z0 += h2[j] * o3w[j * 2];
    z1 += h2[j] * o3w[j * 2 + 1];
  }
  float mx = fmaxf(z0, z1);
  float lse = mx + logf(__expf(z0 - mx) + __expf(z1 - mx));
  float u0 = u[(size_t)t * 2], u1 = u[(size_t)t * 2 + 1];
  float g0 = -logf(-logf(u0 + 1e-10f) + 1e-10f);
  float g1 = -logf(-logf(u1 + 1e-10f) + 1e-10f);
  float s0 = (z0 - lse) + g0;
  float s1 = (z1 - lse) + g1;
  pol[t] = (s0 >= s1) ? 1.0f : 0.0f;  // argmax, ties -> class 0 (jnp.argmax)
}

// ---------------------------------------------------------------------------
// Flash attention, policy-masked. One wave owns a 16-query tile of one (b,h).
// N=2304, dh=64, scale=1/8. Online softmax; score tiles staged in per-wave
// LDS for the C-fragment -> A-fragment relayout between QK^T and P@V.
// qkv layout per token: [q(2x64) | k(2x64) | v(2x64)] = 384 floats.
// kp[q][k] = pol[k] + (1-pol[k]) * (k==q).
// grid = 144 blocks x 256 threads (4 b * 2 h * 144 qtiles = 1152 waves).
// ---------------------------------------------------------------------------
__global__ void __launch_bounds__(256)
attn_kernel(const float* __restrict__ qkv, const float* __restrict__ policy,
            float* __restrict__ out, int polOff) {
  const int SL = 34;  // padded row stride to dodge bank conflicts
  __shared__ float lds[8][16 * 34 + 48];
  int wave = threadIdx.x >> 5, lane = threadIdx.x & 31;
  int lh = lane >> 4, l16 = lane & 15;
  int task = blockIdx.x * 8 + wave;
  int qt = task % 144, bh = task / 144;
  int h = bh & 1, b = bh >> 1;
  int qbase = qt * 16;
  float* S = &lds[wave][0];
  float* rowAlpha = &lds[wave][16 * SL];
  float* rowLinv = rowAlpha + 16;
  const float* pol = policy + b * 4608 + polOff;
  const float* base = qkv + (size_t)b * 2304 * 384;

  // Preload Q as two A-fragments (d = 0..31 and 32..63), bf16.
  v16bf qa0, qa1;
  {
    const float* qp = base + (size_t)(qbase + l16) * 384 + h * 64;
#pragma unroll
    for (int j = 0; j < 8; ++j) {
      int k = 2 * j + (lh << 4);
      qa0[2 * j] = (bf16)qp[k];
      qa0[2 * j + 1] = (bf16)qp[k + 1];
      qa1[2 * j] = (bf16)qp[32 + k];
      qa1[2 * j + 1] = (bf16)qp[32 + k + 1];
    }
  }
  v8f oacc[4];
#pragma unroll
  for (int i = 0; i < 4; ++i)
#pragma unroll
    for (int r = 0; r < 8; ++r) oacc[i][r] = 0.f;
  float mrow = -1e30f, lrow = 0.0f;  // row stats live on lanes 0..15 (row = lane)

  for (int kt = 0; kt < 72; ++kt) {
    int kb = kt * 32;
    // ---- S = (Q K^T) * 0.125 for 32 keys, via 2 sub-tiles x 2 k-steps ----
#pragma unroll
    for (int sub = 0; sub < 2; ++sub) {
      const float* kp0 = base + (size_t)(kb + sub * 16 + l16) * 384 + 128 + h * 64;
      v16bf b0, b1;
#pragma unroll
      for (int j = 0; j < 8; ++j) {
        int d = 2 * j + (lh << 4);
        b0[2 * j] = (bf16)kp0[d];
        b0[2 * j + 1] = (bf16)kp0[d + 1];
        b1[2 * j] = (bf16)kp0[32 + d];
        b1[2 * j + 1] = (bf16)kp0[32 + d + 1];
      }
      v8f s_ = {0.f, 0.f, 0.f, 0.f, 0.f, 0.f, 0.f, 0.f};
      s_ = WMMA_BF16(qa0, b0, s_);
      s_ = WMMA_BF16(qa1, b1, s_);
#pragma unroll
      for (int r = 0; r < 8; ++r)
        S[(r + lh * 8) * SL + sub * 16 + l16] = s_[r] * 0.125f;
    }
    asm volatile("s_wait_dscnt 0" ::: "memory");  // wave-local LDS visibility
    // ---- online softmax row update (lanes 0..15 own one row each) ----
    if (lane < 16) {
      int row = lane, qg = qbase + row;
      float tmax = -1e30f;
      for (int j = 0; j < 32; ++j) tmax = fmaxf(tmax, S[row * SL + j]);
      float mnew = fmaxf(mrow, tmax);
      float alpha = __expf(mrow - mnew);
      float es = 0.f;
      for (int j = 0; j < 32; ++j) {
        int kg = kb + j;
        float kpv = (kg == qg) ? 1.0f : pol[kg];
        float e = __expf(S[row * SL + j] - mnew) * kpv;
        S[row * SL + j] = e;
        es += e;
      }
      lrow = lrow * alpha + es;
      mrow = mnew;
      rowAlpha[row] = alpha;
    }
    asm volatile("s_wait_dscnt 0" ::: "memory");
    // ---- rescale accumulators, then o += P @ V (P is 16x32 A-fragment) ----
#pragma unroll
    for (int r = 0; r < 8; ++r) {
      float al = rowAlpha[r + lh * 8];
      oacc[0][r] *= al;
      oacc[1][r] *= al;
      oacc[2][r] *= al;
      oacc[3][r] *= al;
    }
    v16bf pa;
#pragma unroll
    for (int j = 0; j < 8; ++j) {
      int k = 2 * j + (lh << 4);
      pa[2 * j] = (bf16)S[l16 * SL + k];
      pa[2 * j + 1] = (bf16)S[l16 * SL + k + 1];
    }
#pragma unroll
    for (int nt = 0; nt < 4; ++nt) {
      v16bf vb;
      int vcol = nt * 16 + l16;
#pragma unroll
      for (int j = 0; j < 8; ++j) {
        int k = 2 * j + (lh << 4);
        const float* vp = base + (size_t)(kb + k) * 384 + 256 + h * 64;
        vb[2 * j] = (bf16)vp[vcol];
        vb[2 * j + 1] = (bf16)vp[384 + vcol];
      }
      oacc[nt] = WMMA_BF16(pa, vb, oacc[nt]);
    }
  }
  // ---- finalize: o / (l + 1e-6), write [b, q, h*64 + d] ----
  if (lane < 16) rowLinv[lane] = 1.0f / (lrow + 1e-6f);
  asm volatile("s_wait_dscnt 0" ::: "memory");
#pragma unroll
  for (int r = 0; r < 8; ++r) {
    int m = r + lh * 8;
    float li = rowLinv[m];
    float* op = out + (size_t)(b * 2304 + qbase + m) * 128 + h * 64;
    op[l16] = oacc[0][r] * li;
    op[16 + l16] = oacc[1][r] * li;
    op[32 + l16] = oacc[2][r] * li;
    op[48 + l16] = oacc[3][r] * li;
  }
}

// ---------------------------------------------------------------------------
// Head: lr/ld = x @ head_w(128x20) + head_b, ens = softmax(alpha) blend.
// d_out layout: [3, B*N, 20].
// ---------------------------------------------------------------------------
__global__ void head_kernel(const float* __restrict__ xr, const float* __restrict__ xd,
                            const float* __restrict__ hw, const float* __restrict__ hb,
                            const float* __restrict__ alpha, float* __restrict__ out) {
  int i = blockIdx.x * blockDim.x + threadIdx.x;
  if (i >= 9216 * 20) return;
  int t = i / 20, c = i - t * 20;
  const float* xrr = xr + (size_t)t * 128;
  const float* xdd = xd + (size_t)t * 128;
  float lr = hb[c], ld = hb[c];
  for (int k = 0; k < 128; ++k) {
    float w = hw[k * 20 + c];
    lr += xrr[k] * w;
    ld += xdd[k] * w;
  }
  float a0 = alpha[0], a1 = alpha[1];
  float m = fmaxf(a0, a1);
  float e0 = __expf(a0 - m), e1 = __expf(a1 - m);
  float inv = 1.0f / (e0 + e1);
  out[i] = lr;
  out[184320 + i] = ld;
  out[2 * 184320 + i] = (e0 * inv) * lr + (e1 * inv) * ld;
}

// ---------------------------------------------------------------------------
extern "C" void kernel_launch(void* const* d_in, const int* in_sizes, int n_in,
                              void* d_out, int out_size, void* d_ws, size_t ws_size,
                              hipStream_t stream) {
  (void)in_sizes; (void)n_in; (void)out_size; (void)ws_size;
  const float* rgb = (const float*)d_in[0];
  const float* depth = (const float*)d_in[1];
  const float* u = (const float*)d_in[2];
  const float* p_ln_g = (const float*)d_in[3];
  const float* p_ln_b = (const float*)d_in[4];
  const float* p_in_w = (const float*)d_in[5];
  const float* p_in_b = (const float*)d_in[6];
  const float* p_o1_w = (const float*)d_in[7];
  const float* p_o1_b = (const float*)d_in[8];
  const float* p_o2_w = (const float*)d_in[9];
  const float* p_o2_b = (const float*)d_in[10];
  const float* p_o3_w = (const float*)d_in[11];
  const float* p_o3_b = (const float*)d_in[12];
  const float* head_w = (const float*)d_in[13];
  const float* head_b = (const float*)d_in[14];
  const float* alpha = (const float*)d_in[15];
  // per-stream block params: 0 ln1_g 1 ln1_b 2 qkv_w 3 qkv_b 4 proj_w 5 proj_b
  //                          6 ln2_g 7 ln2_b 8 fc1_w 9 fc1_b 10 fc2_w 11 fc2_b
  const float* blk[2][12];
  for (int s = 0; s < 2; ++s)
    for (int i = 0; i < 12; ++i) blk[s][i] = (const float*)d_in[16 + s * 12 + i];

  char* wsb = (char*)d_ws;
  size_t off = 0;
  auto take = [&](size_t bytes) -> char* {
    char* p = wsb + off;
    off = (off + bytes + 255) & ~(size_t)255;
    return p;
  };
  uint32_t* pkPin = (uint32_t*)take(128 * 128 * 2);
  uint32_t* pkPo1 = (uint32_t*)take(128 * 64 * 2);
  uint32_t *pkQkv[2], *pkProj[2], *pkFc1[2], *pkFc2[2];
  for (int s = 0; s < 2; ++s) {
    pkQkv[s] = (uint32_t*)take(128 * 384 * 2);
    pkProj[s] = (uint32_t*)take(128 * 128 * 2);
    pkFc1[s] = (uint32_t*)take(128 * 512 * 2);
    pkFc2[s] = (uint32_t*)take(512 * 128 * 2);
  }
  float* xlnP = (float*)take((size_t)18432 * 128 * 4);
  float* hP = (float*)take((size_t)18432 * 128 * 4);
  float* glob = (float*)take(256 * 4);
  float* h1 = (float*)take((size_t)18432 * 64 * 4);
  float* pol = (float*)take(18432 * 4);
  float* xln = (float*)take((size_t)9216 * 128 * 4);
  float* qkvb = (float*)take((size_t)9216 * 384 * 4);
  float* attno = (float*)take((size_t)9216 * 128 * 4);
  float* x2 = (float*)take((size_t)9216 * 128 * 4);
  float* hfb = (float*)take((size_t)9216 * 512 * 4);
  float* xout[2] = {(float*)take((size_t)9216 * 128 * 4), (float*)take((size_t)9216 * 128 * 4)};

  auto pk = [&](const float* s, uint32_t* d, int K, int Nc) {
    int total = (K * Nc) / 2;
    pack_w<<<(total + 255) / 256, 256, 0, stream>>>(s, d, K, Nc);
  };
  pk(p_in_w, pkPin, 128, 128);
  pk(p_o1_w, pkPo1, 128, 64);
  for (int s = 0; s < 2; ++s) {
    pk(blk[s][2], pkQkv[s], 128, 384);
    pk(blk[s][4], pkProj[s], 128, 128);
    pk(blk[s][8], pkFc1[s], 128, 512);
    pk(blk[s][10], pkFc2[s], 512, 128);
  }

  // ---- Predictor ----
  ln_kernel<<<18432 / 8, 256, 0, stream>>>(rgb, depth, p_ln_g, p_ln_b, xlnP, 18432, 2304);
  gemm_wmma<<<((18432 / 16) * (128 / 16)) / 8, 256, 0, stream>>>(
      xlnP, pkPin, p_in_b, nullptr, nullptr, hP, 18432, 128, 128, 1, 0);
  glob_reduce<<<4 * 64, 256, 0, stream>>>(hP, glob);
  gemm_wmma<<<((18432 / 16) * (64 / 16)) / 8, 256, 0, stream>>>(
      hP, pkPo1, p_o1_b, nullptr, glob, h1, 18432, 128, 64, 1, 4608);
  pred_tail<<<18432 / 256, 256, 0, stream>>>(h1, p_o2_w, p_o2_b, p_o3_w, p_o3_b, u, pol);

  // ---- Transformer blocks (rgb then depth) ----
  for (int s = 0; s < 2; ++s) {
    const float* x = s ? depth : rgb;
    ln_kernel<<<9216 / 8, 256, 0, stream>>>(x, nullptr, blk[s][0], blk[s][1], xln, 9216, 0);
    gemm_wmma<<<((9216 / 16) * (384 / 16)) / 8, 256, 0, stream>>>(
        xln, pkQkv[s], blk[s][3], nullptr, nullptr, qkvb, 9216, 128, 384, 0, 0);
    attn_kernel<<<144, 256, 0, stream>>>(qkvb, pol, attno, s ? 2304 : 0);
    gemm_wmma<<<((9216 / 16) * (128 / 16)) / 8, 256, 0, stream>>>(
        attno, pkProj[s], blk[s][5], x, nullptr, x2, 9216, 128, 128, 0, 0);
    ln_kernel<<<9216 / 8, 256, 0, stream>>>(x2, nullptr, blk[s][6], blk[s][7], xln, 9216, 0);
    gemm_wmma<<<((9216 / 16) * (512 / 16)) / 8, 256, 0, stream>>>(
        xln, pkFc1[s], blk[s][9], nullptr, nullptr, hfb, 9216, 128, 512, 1, 0);
    gemm_wmma<<<((9216 / 16) * (128 / 16)) / 8, 256, 0, stream>>>(
        hfb, pkFc2[s], blk[s][11], x2, nullptr, xout[s], 9216, 512, 128, 0, 0);
  }

  // ---- Heads + ensemble ----
  head_kernel<<<(9216 * 20 + 255) / 256, 256, 0, stream>>>(
      xout[0], xout[1], head_w, head_b, alpha, (float*)d_out);
}
